// UnifiedMultiHeadTransformerLSTMCell_32538672234979
// MI455X (gfx1250) — matrix operand order repaired and checked by reading, not compiled, not run on hardware
//
#include <hip/hip_runtime.h>

typedef _Float16 half_t;
typedef __attribute__((ext_vector_type(16))) _Float16 v16h;
typedef __attribute__((ext_vector_type(8)))  _Float16 v8h;
typedef __attribute__((ext_vector_type(8)))  float    v8f;
typedef __attribute__((ext_vector_type(4)))  float    v4f;

#define B_   2048
#define I_   1024
#define H_   1024
#define H4_  4096
#define G_   2048          // I + H
#define ML_  32
#define NH_  8
#define D_   128
#define BN_  256
#define BH_  (B_ * H_)

__device__ __forceinline__ float sigm(float x) { return 1.0f / (1.0f + __expf(-x)); }

// ---------------------------------------------------------------------------
// Generic tiled WMMA GEMM:  C[m, coloff+n] = sum_k A[m,k] * W[n,k] + bias[n]
// A: f16 [M, lda], W: f16 [N, ldw] (row-major "weight-transposed" layout),
// C: f32 [M, ldc]. Optional f16 mirror C16 (scaled). M%128==0, N%64==0, K%32==0.
// Block = 256 threads = 8 waves. Block tile 128x64: each wave owns a 16-row
// strip and 4 accumulators (4x 16x16 WMMA per k-step, A-fragment reused).
// Global->LDS uses CDNA5 async-to-LDS (ASYNCcnt) with double buffering, so
// the next K-tile streams in while WMMA consumes the current one.
// Per wave per tile: exactly 3 async instructions (2 A + 1 B, full EXEC),
// so "s_wait_asynccnt 3" after issuing tile t+1 waits precisely for tile t.
// ---------------------------------------------------------------------------
#define APITCH 40
#define BPITCH 40

__device__ __forceinline__ void gemm_issue_tile(
    const half_t* __restrict__ A, int lda,
    const half_t* __restrict__ Wt, int ldw,
    int M0, int N0, int kk, int tid,
    half_t* AsBuf, half_t* BsBuf)
{
#pragma unroll
  for (int j = 0; j < 2; ++j) {              // A: 128x32 halfs = 512 16B chunks
    int idx = tid + j * 256;
    int row = idx >> 2, seg = idx & 3;
    uint32_t lo = (uint32_t)(size_t)&AsBuf[row * APITCH + seg * 8];
    const half_t* g = A + (size_t)(M0 + row) * lda + kk + seg * 8;
    asm volatile("global_load_async_to_lds_b128 %0, %1, off"
                 :: "v"(lo), "v"(g) : "memory");
  }
  {                                           // B: 64x32 halfs = 256 16B chunks
    int row = tid >> 2, seg = tid & 3;
    uint32_t lo = (uint32_t)(size_t)&BsBuf[row * BPITCH + seg * 8];
    const half_t* g = Wt + (size_t)(N0 + row) * ldw + kk + seg * 8;
    asm volatile("global_load_async_to_lds_b128 %0, %1, off"
                 :: "v"(lo), "v"(g) : "memory");
  }
}

__global__ __launch_bounds__(256) void k_gemm(
    const half_t* __restrict__ A, int lda,
    const half_t* __restrict__ Wt, int ldw,
    float* __restrict__ C, int ldc, int coloff,
    const float* __restrict__ bias,
    half_t* __restrict__ C16, float c16scale,
    int N, int K)
{
  __shared__ __align__(16) half_t As[2][128 * APITCH];
  __shared__ __align__(16) half_t Bs[2][64 * BPITCH];
  const int tid  = threadIdx.x;
  const int N0   = blockIdx.x * 64;
  const int M0   = blockIdx.y * 128;
  const int lane = tid & 31;
  const int wv   = tid >> 5;
  const int r16  = lane & 15;
  const int kbase = (lane < 16) ? 0 : 8;    // A frag: K {0..7,16..23} vs {8..15,24..31}
  const int nbase = (lane < 16) ? 0 : 8;    // B frag: K {0..7,16..23} vs {8..15,24..31}

  v8f acc[4];
#pragma unroll
  for (int j = 0; j < 4; ++j)
#pragma unroll
    for (int i = 0; i < 8; ++i) acc[j][i] = 0.0f;

  const int nk = K >> 5;
  gemm_issue_tile(A, lda, Wt, ldw, M0, N0, 0, tid, As[0], Bs[0]);

  for (int t = 0; t < nk; ++t) {
    const int buf = t & 1;
    if (t + 1 < nk) {
      gemm_issue_tile(A, lda, Wt, ldw, M0, N0, (t + 1) << 5, tid,
                      As[buf ^ 1], Bs[buf ^ 1]);
      asm volatile("s_wait_asynccnt 0x3" ::: "memory");
    } else {
      asm volatile("s_wait_asynccnt 0x0" ::: "memory");
    }
    __syncthreads();   // current tile visible to all waves

    const half_t* ap = &As[buf][(wv * 16 + r16) * APITCH];
    v8h a0 = *(const v8h*)(ap + kbase);
    v8h a1 = *(const v8h*)(ap + kbase + 16);
    v16h af;
#pragma unroll
    for (int i = 0; i < 8; ++i) { af[i] = a0[i]; af[i + 8] = a1[i]; }

#pragma unroll
    for (int j = 0; j < 4; ++j) {
      const half_t* bp = &Bs[buf][(j * 16 + r16) * BPITCH + nbase];
      v8h b0 = *(const v8h*)(bp);
      v8h b1 = *(const v8h*)(bp + 16);
      v16h bf;
#pragma unroll
      for (int i = 0; i < 8; ++i) { bf[i] = b0[i]; bf[i + 8] = b1[i]; }
      acc[j] = __builtin_amdgcn_wmma_f32_16x16x32_f16(
          false, af, false, bf, (short)0, acc[j], false, false);
    }
    __syncthreads();   // reads done before this buffer is overwritten (t+2)
  }

  const int mBase = M0 + wv * 16 + ((lane < 16) ? 0 : 8);
#pragma unroll
  for (int j = 0; j < 4; ++j) {
    const int   n  = j * 16 + r16;
    const float bv = bias ? bias[N0 + n] : 0.0f;
#pragma unroll
    for (int r = 0; r < 8; ++r) {
      float  val = acc[j][r] + bv;
      size_t o   = (size_t)(mBase + r) * ldc + coloff + N0 + n;
      if (C)   C[o]   = val;
      if (C16) C16[o] = (half_t)(val * c16scale);
    }
  }
}

// ---------------------------------------------------------------------------
// Attention score contraction: logits[b,h,k] = sum_e U[b,h,e] * K[k,b,e]
// selfShift=1: rows 0..30 come from mem[k+1], row 31 from `last` (h_lstm).
// ---------------------------------------------------------------------------
__global__ __launch_bounds__(256) void k_logits(
    const float* __restrict__ mem, const float* __restrict__ last,
    const float* __restrict__ U, float* __restrict__ logits, int selfShift)
{
  __shared__ float Ks[32][129];   // pad -> conflict-free column reads
  __shared__ float Us[8][128];
  const int b = blockIdx.x;
  const int t = threadIdx.x;
  const int h = t >> 5, k = t & 31;
  float acc = 0.0f;
  for (int cch = 0; cch < 8; ++cch) {
#pragma unroll
    for (int i = 0; i < 16; ++i) {
      int idx = t + i * 256;
      int row = idx >> 7, e = idx & 127;
      const float* rp = (selfShift && row == 31)
          ? (last + (size_t)b * H_)
          : (mem + (size_t)(row + selfShift) * BH_ + (size_t)b * H_);
      Ks[row][e] = __builtin_nontemporal_load(rp + cch * 128 + e);
    }
#pragma unroll
    for (int i = 0; i < 4; ++i) {
      int idx = t + i * 256;
      int hh = idx >> 7, e = idx & 127;
      Us[hh][e] = U[(size_t)b * 8192 + (size_t)hh * 1024 + cch * 128 + e];
    }
    __syncthreads();
#pragma unroll 8
    for (int e = 0; e < 128; ++e) acc += Us[h][e] * Ks[k][e];
    __syncthreads();
  }
  logits[(size_t)b * 256 + h * 32 + k] = acc;
}

__global__ __launch_bounds__(256) void k_softmax32(
    const float* __restrict__ logits, float* __restrict__ att, int nrows)
{
  int wid = threadIdx.x >> 5, lane = threadIdx.x & 31;
  int row = blockIdx.x * 8 + wid;
  if (row >= nrows) return;
  float v = logits[(size_t)row * 32 + lane];
  float m = v;
  for (int off = 16; off > 0; off >>= 1) m = fmaxf(m, __shfl_xor(m, off, 32));
  float e = __expf(v - m);
  float s = e;
  for (int off = 16; off > 0; off >>= 1) s += __shfl_xor(s, off, 32);
  att[(size_t)row * 32 + lane] = e / s;
}

// wsum16[b,h,e] = sum_k att[b,h,k] * V[k,b,e]   (streaming, memory-bound)
__global__ __launch_bounds__(256) void k_wsum(
    const float* __restrict__ memv, const float* __restrict__ last,
    const float* __restrict__ att, half_t* __restrict__ wsum16, int selfShift)
{
  __shared__ float atts[256];
  const int b = blockIdx.x;
  const int t = threadIdx.x;
  atts[t] = att[(size_t)b * 256 + t];
  __syncthreads();
  const int e0 = t * 4;
  float acc[NH_][4];
#pragma unroll
  for (int h = 0; h < NH_; ++h)
#pragma unroll
    for (int j = 0; j < 4; ++j) acc[h][j] = 0.0f;

  for (int k = 0; k < 32; ++k) {
    const float* rp = (selfShift && k == 31)
        ? (last + (size_t)b * H_)
        : (memv + (size_t)(k + selfShift) * BH_ + (size_t)b * H_);
    if (k < 31) {
      const float* np = (selfShift && k == 30)
          ? (last + (size_t)b * H_)
          : (memv + (size_t)(k + 1 + selfShift) * BH_ + (size_t)b * H_);
      __builtin_prefetch(np + e0, 0, 0);
    }
    v4f v = __builtin_nontemporal_load((const v4f*)(rp + e0));
#pragma unroll
    for (int h = 0; h < NH_; ++h) {
      float a = atts[h * 32 + k];
#pragma unroll
      for (int j = 0; j < 4; ++j) acc[h][j] += a * v[j];
    }
  }
#pragma unroll
  for (int h = 0; h < NH_; ++h)
#pragma unroll
    for (int j = 0; j < 4; ++j)
      wsum16[(size_t)b * 8192 + (size_t)h * 1024 + e0 + j] = (half_t)acc[h][j];
}

// temporal_summary = mean over memK_new rows  -> f16
__global__ __launch_bounds__(256) void k_tsum(
    const float* __restrict__ memk, const float* __restrict__ hl,
    half_t* __restrict__ ts16)
{
  const int b  = blockIdx.x;
  const int e0 = threadIdx.x * 4;
  v4f s = *(const v4f*)(hl + (size_t)b * H_ + e0);
  for (int k = 1; k < 32; ++k) {
    v4f v = __builtin_nontemporal_load(
        (const v4f*)(memk + (size_t)k * BH_ + (size_t)b * H_ + e0));
#pragma unroll
    for (int j = 0; j < 4; ++j) s[j] += v[j];
  }
#pragma unroll
  for (int j = 0; j < 4; ++j)
    ts16[(size_t)b * H_ + e0 + j] = (half_t)(s[j] * (1.0f / 32.0f));
}

// router: rw = softmax2(h_lstm @ Wrout^T + brout), one wave per batch row
__global__ __launch_bounds__(256) void k_router(
    const float* __restrict__ hl, const float* __restrict__ wr,
    const float* __restrict__ br, float* __restrict__ rw)
{
  int wid = threadIdx.x >> 5, lane = threadIdx.x & 31;
  int b = blockIdx.x * 8 + wid;
  float a0 = 0.0f, a1 = 0.0f;
  for (int e = lane; e < H_; e += 32) {
    float hv = hl[(size_t)b * H_ + e];
    a0 += hv * wr[e];
    a1 += hv * wr[H_ + e];
  }
  for (int off = 16; off > 0; off >>= 1) {
    a0 += __shfl_xor(a0, off, 32);
    a1 += __shfl_xor(a1, off, 32);
  }
  if (lane == 0) {
    float z0 = a0 + br[0], z1 = a1 + br[1];
    float m = fmaxf(z0, z1);
    float e0 = __expf(z0 - m), e1 = __expf(z1 - m);
    float inv = 1.0f / (e0 + e1);
    rw[(size_t)b * 2]     = e0 * inv;
    rw[(size_t)b * 2 + 1] = e1 * inv;
  }
}

__global__ __launch_bounds__(256) void k_lstm(
    const float* __restrict__ gates, const float* __restrict__ c,
    float* __restrict__ h_out, float* __restrict__ c_out,
    half_t* __restrict__ hl16)
{
  size_t idx = (size_t)blockIdx.x * 256 + threadIdx.x;
  int b = (int)(idx >> 10), e = (int)(idx & 1023);
  const float* g = gates + (size_t)b * 4096;
  float ig = sigm(g[e]);
  float fg = sigm(g[1024 + e]);
  float gg = tanhf(g[2048 + e]);
  float og = sigm(g[3072 + e]);
  float cn = fg * c[idx] + ig * gg;
  float hv = og * tanhf(cn);
  c_out[idx] = cn;
  h_out[idx] = hv;
  hl16[idx]  = (half_t)hv;
}

__global__ void k_cast(half_t* __restrict__ dst, const float* __restrict__ src,
                       size_t n, float sc)
{
  size_t i = (size_t)blockIdx.x * 256 + threadIdx.x;
  if (i < n) dst[i] = (half_t)(src[i] * sc);
}

__global__ void k_cast2d(half_t* __restrict__ dst, int ldd, int off,
                         const float* __restrict__ src, int lds_, int R, int C)
{
  size_t i = (size_t)blockIdx.x * 256 + threadIdx.x;
  if (i >= (size_t)R * C) return;
  int r = (int)(i / C), cc = (int)(i % C);
  dst[(size_t)r * ldd + off + cc] = (half_t)src[(size_t)r * lds_ + cc];
}

__global__ void k_transpose(half_t* __restrict__ dst, const float* __restrict__ src,
                            int R, int C)
{
  size_t i = (size_t)blockIdx.x * 256 + threadIdx.x;
  if (i >= (size_t)R * C) return;
  int r = (int)(i / C), cc = (int)(i % C);
  dst[(size_t)cc * R + r] = (half_t)src[i];
}

__global__ void k_addvec(float* __restrict__ d, const float* __restrict__ a,
                         const float* __restrict__ b, int n)
{
  int i = blockIdx.x * 256 + threadIdx.x;
  if (i < n) d[i] = a[i] + b[i];
}

__global__ void k_ra(const float* __restrict__ rw, const float* __restrict__ so,
                     const float* __restrict__ eo, float* __restrict__ ra,
                     half_t* __restrict__ ra16)
{
  size_t i = (size_t)blockIdx.x * 256 + threadIdx.x;
  int b = (int)(i >> 10);
  float v = rw[2 * b] * so[i] + rw[2 * b + 1] * eo[i];
  ra[i]   = v;
  ra16[i] = (half_t)v;
}

__global__ void k_routed(const float* __restrict__ hl, const float* __restrict__ ra,
                         const float* __restrict__ res, float* __restrict__ routed,
                         half_t* __restrict__ cg16)
{
  size_t i = (size_t)blockIdx.x * 256 + threadIdx.x;
  int b = (int)(i >> 10), e = (int)(i & 1023);
  float v = hl[i] + ra[i] + res[i];
  routed[i] = v;
  cg16[(size_t)b * G_ + I_ + e] = (half_t)v;
}

__global__ void k_final(const float* __restrict__ hl, const float* __restrict__ routed,
                        const float* __restrict__ comp, const float* __restrict__ gz,
                        const float* __restrict__ expd, float* __restrict__ outp)
{
  size_t i = (size_t)blockIdx.x * 256 + threadIdx.x;
  int b = (int)(i >> 10), e = (int)(i & 1023);
  float ce  = comp[(size_t)b * BN_ + (e & (BN_ - 1))];
  float g   = sigm(gz[i]);
  float hct = ce + g * (expd[i] - ce);
  outp[i] = (hl[i] + routed[i] + hct) * (1.0f / 3.0f);
}

// ---------------------------------------------------------------------------
extern "C" void kernel_launch(void* const* d_in, const int* in_sizes, int n_in,
                              void* d_out, int out_size, void* d_ws, size_t ws_size,
                              hipStream_t stream)
{
  (void)in_sizes; (void)n_in; (void)out_size; (void)ws_size;

  const float* x     = (const float*)d_in[0];
  const float* h     = (const float*)d_in[1];
  const float* c     = (const float*)d_in[2];
  const float* memK  = (const float*)d_in[3];
  const float* memV  = (const float*)d_in[4];
  const float* extK  = (const float*)d_in[5];
  const float* extV  = (const float*)d_in[6];
  const float* W_ih  = (const float*)d_in[7];
  const float* W_hh  = (const float*)d_in[8];
  const float* b_ih  = (const float*)d_in[9];
  const float* b_hh  = (const float*)d_in[10];
  const float* sWq   = (const float*)d_in[11]; const float* sbq = (const float*)d_in[12];
  const float* sWk   = (const float*)d_in[13]; /* sbk cancels in softmax */
  const float* sWv   = (const float*)d_in[15]; const float* sbv = (const float*)d_in[16];
  const float* sWo   = (const float*)d_in[17]; const float* sbo = (const float*)d_in[18];
  const float* cWq   = (const float*)d_in[19]; const float* cbq = (const float*)d_in[20];
  const float* cWk   = (const float*)d_in[21]; /* cbk cancels in softmax */
  const float* cWv   = (const float*)d_in[23]; const float* cbv = (const float*)d_in[24];
  const float* cWo   = (const float*)d_in[25]; const float* cbo = (const float*)d_in[26];
  const float* Wres  = (const float*)d_in[27]; const float* bres  = (const float*)d_in[28];
  const float* Wrout = (const float*)d_in[29]; const float* brout = (const float*)d_in[30];
  const float* Wcomp = (const float*)d_in[31]; const float* bcomp = (const float*)d_in[32];
  const float* Wexp  = (const float*)d_in[33]; const float* bexp  = (const float*)d_in[34];
  const float* Wgate = (const float*)d_in[35]; const float* bgate = (const float*)d_in[36];

  float* out    = (float*)d_out;
  float* hFinal = out;                 // [B,H]
  float* h_lstm = out + (size_t)BH_;   // [B,H]
  float* c_next = out + 2 * (size_t)BH_;

  // --- workspace bump allocator ---
  char*  base = (char*)d_ws;
  size_t off  = 0;
  auto allocB = [&](size_t bytes) -> void* {
    off = (off + 255) & ~(size_t)255;
    void* p = base + off;
    off += bytes;
    return p;
  };
  auto allocH = [&](size_t n) { return (half_t*)allocB(n * sizeof(half_t)); };
  auto allocF = [&](size_t n) { return (float*)allocB(n * sizeof(float)); };

  half_t* cxh16   = allocH((size_t)B_ * G_);
  half_t* Wihh16  = allocH((size_t)H4_ * G_);
  half_t* sWq16   = allocH((size_t)H_ * H_);
  half_t* sWv16   = allocH((size_t)H_ * H_);
  half_t* sWo16   = allocH((size_t)H_ * H_);
  half_t* cWq16   = allocH((size_t)H_ * H_);
  half_t* cWv16   = allocH((size_t)H_ * H_);
  half_t* cWo16   = allocH((size_t)H_ * H_);
  half_t* sWkT16  = allocH((size_t)H_ * H_);
  half_t* cWkT16  = allocH((size_t)H_ * H_);
  half_t* Wres16  = allocH((size_t)H_ * H_);
  half_t* Wcomp16 = allocH((size_t)BN_ * H_);
  half_t* Wexp16  = allocH((size_t)H_ * BN_);
  half_t* Wgate16 = allocH((size_t)H_ * G_);
  half_t* hl16    = allocH((size_t)BH_);
  half_t* qs16    = allocH((size_t)BH_);
  half_t* qc16    = allocH((size_t)BH_);
  half_t* wsumS16 = allocH((size_t)B_ * 8192);
  half_t* wsumE16 = allocH((size_t)B_ * 8192);
  half_t* hoS16   = allocH((size_t)BH_);
  half_t* hoE16   = allocH((size_t)BH_);
  half_t* ra16    = allocH((size_t)BH_);
  half_t* ts16    = allocH((size_t)BH_);
  half_t* comp16  = allocH((size_t)B_ * BN_);
  half_t* cg16    = allocH((size_t)B_ * G_);

  float* bihh    = allocF((size_t)H4_);
  float* gates   = allocF((size_t)B_ * H4_);
  float* tmpq    = allocF((size_t)BH_);
  float* Uself   = allocF((size_t)B_ * 8192);
  float* Uext    = allocF((size_t)B_ * 8192);
  float* logS    = allocF((size_t)B_ * 256);
  float* logE    = allocF((size_t)B_ * 256);
  float* attS    = allocF((size_t)B_ * 256);
  float* attE    = allocF((size_t)B_ * 256);
  float* headoS  = allocF((size_t)BH_);
  float* headoE  = allocF((size_t)BH_);
  float* selfO   = allocF((size_t)BH_);
  float* extO    = allocF((size_t)BH_);
  float* raf     = allocF((size_t)BH_);
  float* restmp  = allocF((size_t)BH_);
  float* routedf = allocF((size_t)BH_);
  float* comp    = allocF((size_t)B_ * BN_);
  float* expd    = allocF((size_t)BH_);
  float* gatez   = allocF((size_t)BH_);
  float* rw      = allocF((size_t)B_ * 2);

  auto cast1 = [&](half_t* dst, const float* src, size_t n, float sc) {
    k_cast<<<dim3((unsigned)((n + 255) / 256)), 256, 0, stream>>>(dst, src, n, sc);
  };
  auto cast2d = [&](half_t* dst, int ldd, int o2, const float* src, int lds_, int R, int C) {
    k_cast2d<<<dim3((unsigned)(((size_t)R * C + 255) / 256)), 256, 0, stream>>>(
        dst, ldd, o2, src, lds_, R, C);
  };
  auto gemm = [&](const half_t* A, int lda, const half_t* Wm, int ldw,
                  float* C, int ldc, int coloff, const float* bias,
                  half_t* C16, float c16scale, int M, int N, int K) {
    dim3 g((unsigned)(N / 64), (unsigned)(M / 128));
    k_gemm<<<g, 256, 0, stream>>>(A, lda, Wm, ldw, C, ldc, coloff, bias, C16, c16scale, N, K);
  };

  // ---- precision conversions / layout builds ----
  cast2d(cxh16, G_, 0,   x, I_, B_, I_);
  cast2d(cxh16, G_, I_,  h, H_, B_, H_);
  cast2d(Wihh16, G_, 0,  W_ih, I_, H4_, I_);
  cast2d(Wihh16, G_, I_, W_hh, H_, H4_, H_);
  k_addvec<<<dim3(16), 256, 0, stream>>>(bihh, b_ih, b_hh, H4_);
  cast1(sWq16, sWq, (size_t)H_ * H_, 1.0f);
  cast1(sWv16, sWv, (size_t)H_ * H_, 1.0f);
  cast1(sWo16, sWo, (size_t)H_ * H_, 1.0f);
  cast1(cWq16, cWq, (size_t)H_ * H_, 1.0f);
  cast1(cWv16, cWv, (size_t)H_ * H_, 1.0f);
  cast1(cWo16, cWo, (size_t)H_ * H_, 1.0f);
  cast1(Wres16, Wres, (size_t)H_ * H_, 1.0f);
  cast1(Wcomp16, Wcomp, (size_t)BN_ * H_, 1.0f);
  cast1(Wexp16, Wexp, (size_t)H_ * BN_, 1.0f);
  cast1(Wgate16, Wgate, (size_t)H_ * G_, 1.0f);
  k_transpose<<<dim3((H_ * H_) / 256), 256, 0, stream>>>(sWkT16, sWk, H_, H_);
  k_transpose<<<dim3((H_ * H_) / 256), 256, 0, stream>>>(cWkT16, cWk, H_, H_);
  cast2d(cg16, G_, 0, x, I_, B_, I_);   // gate-input concat: x half

  // ---- LSTM cell: fused [x|h] @ [W_ih|W_hh]^T ----
  gemm(cxh16, G_, Wihh16, G_, gates, H4_, 0, bihh, nullptr, 1.0f, B_, H4_, G_);
  k_lstm<<<dim3(BH_ / 256), 256, 0, stream>>>(gates, c, h_lstm, c_next, hl16);

  // ---- router + temporal summary (depend only on h_lstm / memK) ----
  k_router<<<dim3(B_ / 8), 256, 0, stream>>>(h_lstm, Wrout, brout, rw);
  k_tsum<<<dim3(B_), 256, 0, stream>>>(memK, h_lstm, ts16);

  const float qscale = 0.0883883476483184f;  // 1/sqrt(128)

  // ---- self attention (reassociated) ----
  gemm(hl16, H_, sWq16, H_, tmpq, H_, 0, sbq, qs16, qscale, B_, H_, H_);
  for (int hh = 0; hh < NH_; ++hh)
    gemm(qs16 + hh * D_, H_, sWkT16 + hh * D_, H_, Uself, 8192, hh * 1024,
         nullptr, nullptr, 1.0f, B_, H_, D_);
  k_logits<<<dim3(B_), 256, 0, stream>>>(memK, h_lstm, Uself, logS, 1);
  k_softmax32<<<dim3((B_ * NH_) / 8), 256, 0, stream>>>(logS, attS, B_ * NH_);
  k_wsum<<<dim3(B_), 256, 0, stream>>>(memV, h_lstm, attS, wsumS16, 1);
  for (int hh = 0; hh < NH_; ++hh)
    gemm(wsumS16 + hh * 1024, 8192, sWv16 + (size_t)hh * D_ * H_, H_,
         headoS, H_, hh * D_, sbv + hh * D_, hoS16, 1.0f, B_, D_, H_);
  gemm(hoS16, H_, sWo16, H_, selfO, H_, 0, sbo, nullptr, 1.0f, B_, H_, H_);

  // ---- external attention (same structure, no row shift) ----
  gemm(hl16, H_, cWq16, H_, tmpq, H_, 0, cbq, qc16, qscale, B_, H_, H_);
  for (int hh = 0; hh < NH_; ++hh)
    gemm(qc16 + hh * D_, H_, cWkT16 + hh * D_, H_, Uext, 8192, hh * 1024,
         nullptr, nullptr, 1.0f, B_, H_, D_);
  k_logits<<<dim3(B_), 256, 0, stream>>>(extK, nullptr, Uext, logE, 0);
  k_softmax32<<<dim3((B_ * NH_) / 8), 256, 0, stream>>>(logE, attE, B_ * NH_);
  k_wsum<<<dim3(B_), 256, 0, stream>>>(extV, nullptr, attE, wsumE16, 0);
  for (int hh = 0; hh < NH_; ++hh)
    gemm(wsumE16 + hh * 1024, 8192, cWv16 + (size_t)hh * D_ * H_, H_,
         headoE, H_, hh * D_, cbv + hh * D_, hoE16, 1.0f, B_, D_, H_);
  gemm(hoE16, H_, cWo16, H_, extO, H_, 0, cbo, nullptr, 1.0f, B_, H_, H_);

  // ---- routing, residual, compress/expand, gating, final mix ----
  k_ra<<<dim3(BH_ / 256), 256, 0, stream>>>(rw, selfO, extO, raf, ra16);
  gemm(ra16, H_, Wres16, H_, restmp, H_, 0, bres, nullptr, 1.0f, B_, H_, H_);
  k_routed<<<dim3(BH_ / 256), 256, 0, stream>>>(h_lstm, raf, restmp, routedf, cg16);
  gemm(ts16, H_, Wcomp16, H_, comp, BN_, 0, bcomp, comp16, 1.0f, B_, BN_, H_);
  gemm(comp16, BN_, Wexp16, BN_, expd, H_, 0, bexp, nullptr, 1.0f, B_, H_, BN_);
  gemm(cg16, G_, Wgate16, G_, gatez, H_, 0, bgate, nullptr, 1.0f, B_, H_, G_);
  k_final<<<dim3(BH_ / 256), 256, 0, stream>>>(h_lstm, routedf, comp, gatez, expd, hFinal);
}
